// LightweightPointCloudEncoderWithGlobalAttention_86268713107572
// MI455X (gfx1250) — matrix operand order repaired and checked by reading, not compiled, not run on hardware
//
#include <hip/hip_runtime.h>
#include <hip/hip_bf16.h>

#define NPTS 4096
#define BATCH 4
#define CIN 9
#define KNN 8
#define USE_ASYNC_LDS 1

typedef __attribute__((ext_vector_type(16))) _Float16 v16h;
typedef __attribute__((ext_vector_type(8)))  _Float16 v8h;
typedef __attribute__((ext_vector_type(8)))  float    v8f;

static __device__ __forceinline__ v16h make_frag(const _Float16* p0, const _Float16* p1) {
    v8h a = *(const v8h*)p0;
    v8h b = *(const v8h*)p1;
    v16h r;
#pragma unroll
    for (int t = 0; t < 8; ++t) { r[t] = a[t]; r[t + 8] = b[t]; }
    return r;
}

// 16-byte copy global->LDS, async on CDNA5 (tracked by ASYNCcnt)
static __device__ __forceinline__ void cp16(_Float16* l, const _Float16* g) {
#if USE_ASYNC_LDS
    unsigned laddr = (unsigned)(size_t)l;   // LDS aperture: addr[31:0] is the LDS offset
    asm volatile("global_load_async_to_lds_b128 %0, %1, off" :: "v"(laddr), "v"(g) : "memory");
#else
    *(v8h*)l = *(const v8h*)g;
#endif
}
static __device__ __forceinline__ void cp_commit() {
#if USE_ASYNC_LDS
    asm volatile("s_wait_asynccnt 0x0" ::: "memory");
#endif
}

// ---------------------------------------------------------------------------
// 1) kNN (k=8) + graph feature: feat[b,0:9,n]=mean(neigh)-x, feat[b,9:18,n]=x
// ---------------------------------------------------------------------------
__global__ void knn_feat_kernel(const float* __restrict__ x, float* __restrict__ feat) {
    const int b = blockIdx.y;
    const int i = blockIdx.x * 256 + threadIdx.x;
    const float* xb = x + (size_t)b * NPTS * CIN;
    const float px = xb[i * CIN + 0], py = xb[i * CIN + 1], pz = xb[i * CIN + 2];

    float bd[KNN]; int bi[KNN];
#pragma unroll
    for (int k = 0; k < KNN; ++k) { bd[k] = 3.4e38f; bi[k] = 0; }

    __shared__ float sx[1024], sy[1024], sz[1024];
    for (int base = 0; base < NPTS; base += 1024) {
        __syncthreads();
        for (int t = threadIdx.x; t < 1024; t += 256) {
            sx[t] = xb[(base + t) * CIN + 0];
            sy[t] = xb[(base + t) * CIN + 1];
            sz[t] = xb[(base + t) * CIN + 2];
        }
        __syncthreads();
        for (int jj = 0; jj < 1024; ++jj) {
            const int j = base + jj;
            const float dx = px - sx[jj], dy = py - sy[jj], dz = pz - sz[jj];
            const float d2 = dx * dx + dy * dy + dz * dz;
            if (j != i && d2 < bd[KNN - 1]) {
                bd[KNN - 1] = d2; bi[KNN - 1] = j;
#pragma unroll
                for (int k = KNN - 1; k > 0; --k) {
                    if (bd[k] < bd[k - 1]) {
                        float td = bd[k]; bd[k] = bd[k - 1]; bd[k - 1] = td;
                        int   ti = bi[k]; bi[k] = bi[k - 1]; bi[k - 1] = ti;
                    }
                }
            }
        }
    }

    float mean[CIN];
#pragma unroll
    for (int c = 0; c < CIN; ++c) mean[c] = 0.f;
#pragma unroll
    for (int k = 0; k < KNN; ++k) {
        const float* nb = xb + (size_t)bi[k] * CIN;
#pragma unroll
        for (int c = 0; c < CIN; ++c) mean[c] += nb[c];
    }
    float* fb = feat + (size_t)b * (2 * CIN) * NPTS;
#pragma unroll
    for (int c = 0; c < CIN; ++c) {
        const float xc = xb[i * CIN + c];
        fb[(size_t)c * NPTS + i] = mean[c] * (1.f / KNN) - xc;
        fb[(size_t)(CIN + c) * NPTS + i] = xc;
    }
}

// ---------------------------------------------------------------------------
// 2) WMMA GEMM: Y[b][o][n] = sum_c W[o][c]*X[b][c][n] + bias[o]
//    grid (N/64, CO/32, B), block 128: 4 waves, tile 32(o) x 64(n)
//    h16mode: 0 = f32 out (+Res,+relu), 1 = f16 transposed [n][CO], 2 = f16 [o][n]
// ---------------------------------------------------------------------------
template<int CI, int CO>
__global__ __launch_bounds__(128)
void gemm_kernel(const float* __restrict__ W, const float* __restrict__ X,
                 const float* __restrict__ bias, const float* __restrict__ Res,
                 float* __restrict__ Y, _Float16* __restrict__ Yh,
                 int h16mode, int do_relu) {
    const int n0 = blockIdx.x * 64;
    const int o0 = blockIdx.y * 32;
    const int b  = blockIdx.z;
    const int tid = threadIdx.x;
    const int wv = tid >> 5, lane = tid & 31;
    const int mrow = lane & 15;
    const bool lo = lane < 16;

    __shared__ _Float16 As[32][40];
    __shared__ _Float16 Bs[64][40];

    const float* Xb = X + (size_t)b * CI * NPTS;
    v8f acc0 = {}, acc1 = {};
    constexpr int KC = (CI + 31) / 32;

    for (int kc = 0; kc < KC; ++kc) {
        __syncthreads();
        // stage A: 32 rows x 32 k (each thread: 8 consecutive k of one row)
        {
            const int r = tid >> 2, kq = (tid & 3) * 8;
            v8h pk;
            if constexpr (CI % 32 == 0) {
                const float* src = &W[(size_t)(o0 + r) * CI + kc * 32 + kq];
                const float4 f0 = *(const float4*)(src);
                const float4 f1 = *(const float4*)(src + 4);
                pk[0] = (_Float16)f0.x; pk[1] = (_Float16)f0.y;
                pk[2] = (_Float16)f0.z; pk[3] = (_Float16)f0.w;
                pk[4] = (_Float16)f1.x; pk[5] = (_Float16)f1.y;
                pk[6] = (_Float16)f1.z; pk[7] = (_Float16)f1.w;
            } else {
#pragma unroll
                for (int u = 0; u < 8; ++u) {
                    const int c = kc * 32 + kq + u;
                    pk[u] = (_Float16)((c < CI) ? W[(size_t)(o0 + r) * CI + c] : 0.f);
                }
            }
            *(v8h*)&As[r][kq] = pk;
        }
        // stage B (transpose [c][n] -> [j][k]): each pass: float4 along n
#pragma unroll
        for (int p = 0; p < 4; ++p) {
            const int k  = p * 8 + (tid >> 4);
            const int j0 = (tid & 15) * 4;
            const int c  = kc * 32 + k;
            float4 vsrc;
            if (CI % 32 == 0 || c < CI) vsrc = *(const float4*)&Xb[(size_t)c * NPTS + n0 + j0];
            else                        vsrc = make_float4(0.f, 0.f, 0.f, 0.f);
            Bs[j0 + 0][k] = (_Float16)vsrc.x;
            Bs[j0 + 1][k] = (_Float16)vsrc.y;
            Bs[j0 + 2][k] = (_Float16)vsrc.z;
            Bs[j0 + 3][k] = (_Float16)vsrc.w;
        }
        __syncthreads();

        const _Float16* ap0 = &As[mrow][0];
        const _Float16* ap1 = &As[16 + mrow][0];
        const v16h a0 = lo ? make_frag(ap0, ap0 + 16) : make_frag(ap0 + 8, ap0 + 24);
        const v16h a1 = lo ? make_frag(ap1, ap1 + 16) : make_frag(ap1 + 8, ap1 + 24);
        const _Float16* bp = &Bs[wv * 16 + mrow][lo ? 0 : 8];
        const v16h bf = make_frag(bp, bp + 16);
        acc0 = __builtin_amdgcn_wmma_f32_16x16x32_f16(false, a0, false, bf, (short)0, acc0, false, false);
        acc1 = __builtin_amdgcn_wmma_f32_16x16x32_f16(false, a1, false, bf, (short)0, acc1, false, false);
    }

    const int half8 = lo ? 0 : 8;
    const int n = n0 + wv * 16 + mrow;
#pragma unroll
    for (int t = 0; t < 2; ++t) {
        const v8f a = t ? acc1 : acc0;
        const int ob = o0 + t * 16 + half8;
        if (h16mode == 1) {           // transposed f16: Yh[b][n][o], contiguous o
            v8h pk;
#pragma unroll
            for (int r = 0; r < 8; ++r) pk[r] = (_Float16)(a[r] + bias[ob + r]);
            *(v8h*)&Yh[((size_t)b * NPTS + n) * CO + ob] = pk;
        } else if (h16mode == 2) {    // normal f16: Yh[b][o][n]
#pragma unroll
            for (int r = 0; r < 8; ++r)
                Yh[(size_t)b * CO * NPTS + (size_t)(ob + r) * NPTS + n] = (_Float16)(a[r] + bias[ob + r]);
        } else {
#pragma unroll
            for (int r = 0; r < 8; ++r) {
                const int o = ob + r;
                float v = a[r] + bias[o];
                if (Res) v += Res[(size_t)b * CO * NPTS + (size_t)o * NPTS + n];
                if (do_relu) v = fmaxf(v, 0.f);
                Y[(size_t)b * CO * NPTS + (size_t)o * NPTS + n] = v;
            }
        }
    }
}

// ---------------------------------------------------------------------------
// 3) BatchNorm stats (mean/var over B*N per channel) -> scale/shift; apply
// ---------------------------------------------------------------------------
__global__ void bn_stats_kernel(const float* __restrict__ Y, const float* __restrict__ g,
                                const float* __restrict__ be, int Co,
                                float* __restrict__ scale, float* __restrict__ shift) {
    const int c = blockIdx.x;
    __shared__ float ssum[256], ssq[256];
    float s = 0.f, q = 0.f;
    for (int b = 0; b < BATCH; ++b) {
        const float* row = Y + (size_t)b * Co * NPTS + (size_t)c * NPTS;
        for (int n = threadIdx.x; n < NPTS; n += 256) { const float v = row[n]; s += v; q += v * v; }
    }
    ssum[threadIdx.x] = s; ssq[threadIdx.x] = q;
    __syncthreads();
    for (int st = 128; st > 0; st >>= 1) {
        if (threadIdx.x < st) { ssum[threadIdx.x] += ssum[threadIdx.x + st]; ssq[threadIdx.x] += ssq[threadIdx.x + st]; }
        __syncthreads();
    }
    if (threadIdx.x == 0) {
        const float inv = 1.f / (float)(BATCH * NPTS);
        const float m = ssum[0] * inv;
        const float var = ssq[0] * inv - m * m;
        const float sc = g[c] * rsqrtf(var + 1e-5f);
        scale[c] = sc;
        shift[c] = be[c] - m * sc;
    }
}

__global__ void bn_apply_relu_kernel(float* __restrict__ Y, const float* __restrict__ scale,
                                     const float* __restrict__ shift, int Co) {
    const size_t idx = (size_t)blockIdx.x * 256 + threadIdx.x;
    const int c = (int)((idx / NPTS) % Co);
    Y[idx] = fmaxf(Y[idx] * scale[c] + shift[c], 0.f);
}

// ---------------------------------------------------------------------------
// 4) Flash attention, double-buffered async LDS staging.
//    Qt,Kt: f16 [b][n][128]; V: f16 [b][128][n]. O: f32 [b][128][n]
//    grid (N/64, B), block 128 (wave w -> query rows n0+16w..+15)
// ---------------------------------------------------------------------------
__global__ __launch_bounds__(128)
void flash_attn_kernel(const _Float16* __restrict__ Qt, const _Float16* __restrict__ Kt,
                       const _Float16* __restrict__ Vh, float* __restrict__ Og) {
    const int n0 = blockIdx.x * 64;
    const int b  = blockIdx.y;
    const int tid = threadIdx.x, wv = tid >> 5, lane = tid & 31;
    const int mrow = lane & 15;
    const bool lo = lane < 16;
    const int HD = 128;

    __shared__ _Float16 Ks[2][32][136];
    __shared__ _Float16 Vs[2][128][40];
    __shared__ _Float16 Ps[4][16][40];

    const _Float16* Qb = Qt + (size_t)b * NPTS * HD;
    const _Float16* Kb = Kt + (size_t)b * NPTS * HD;
    const _Float16* Vb = Vh + (size_t)b * HD * NPTS;

    // Q fragments: direct from global (row = contiguous head dim)
    v16h qf[4];
    {
        const _Float16* qrow = Qb + (size_t)(n0 + wv * 16 + mrow) * HD;
#pragma unroll
        for (int kc = 0; kc < 4; ++kc) {
            const int base = kc * 32 + (lo ? 0 : 8);
            qf[kc] = make_frag(qrow + base, qrow + base + 16);
        }
    }

    auto stage = [&](int buf, int m0) {
#pragma unroll
        for (int s = tid; s < 32 * 16; s += 128) {
            const int m = s >> 4, seg = s & 15;
            cp16(&Ks[buf][m][seg * 8], Kb + (size_t)(m0 + m) * HD + seg * 8);
        }
#pragma unroll
        for (int s = tid; s < 128 * 4; s += 128) {
            const int h = s >> 2, seg = s & 3;
            cp16(&Vs[buf][h][seg * 8], Vb + (size_t)h * NPTS + m0 + seg * 8);
        }
    };

    v8f oacc[8];
#pragma unroll
    for (int hc = 0; hc < 8; ++hc)
#pragma unroll
        for (int r = 0; r < 8; ++r) oacc[hc][r] = 0.f;

    float rmax[8], rsum[8];
#pragma unroll
    for (int r = 0; r < 8; ++r) { rmax[r] = -3.0e38f; rsum[r] = 0.f; }

    stage(0, 0);   // prologue: fill buffer 0
    int cur = 0;

    for (int m0 = 0; m0 < NPTS; m0 += 32) {
        cp_commit();        // my wave's copies into Ks/Vs[cur] have landed
        __syncthreads();    // everyone's copies have landed
        if (m0 + 32 < NPTS) stage(cur ^ 1, m0 + 32);   // prefetch next chunk

        v8f s0 = {}, s1 = {};
#pragma unroll
        for (int kc = 0; kc < 4; ++kc) {
            const _Float16* kp0 = &Ks[cur][mrow][kc * 32 + (lo ? 0 : 8)];
            const v16h bf0 = make_frag(kp0, kp0 + 16);
            s0 = __builtin_amdgcn_wmma_f32_16x16x32_f16(false, qf[kc], false, bf0,
                                                        (short)0, s0, false, false);
            const _Float16* kp1 = &Ks[cur][16 + mrow][kc * 32 + (lo ? 0 : 8)];
            const v16h bf1 = make_frag(kp1, kp1 + 16);
            s1 = __builtin_amdgcn_wmma_f32_16x16x32_f16(false, qf[kc], false, bf1,
                                                        (short)0, s1, false, false);
        }

        float alpha[8], p0[8], p1[8];
#pragma unroll
        for (int r = 0; r < 8; ++r) {
            float mx = fmaxf(s0[r], s1[r]);
            mx = fmaxf(mx, __shfl_xor(mx, 1, 32));
            mx = fmaxf(mx, __shfl_xor(mx, 2, 32));
            mx = fmaxf(mx, __shfl_xor(mx, 4, 32));
            mx = fmaxf(mx, __shfl_xor(mx, 8, 32));
            const float nm = fmaxf(rmax[r], mx);
            alpha[r] = __expf(rmax[r] - nm);
            rmax[r] = nm;
            p0[r] = __expf(s0[r] - nm);
            p1[r] = __expf(s1[r] - nm);
            float ps = p0[r] + p1[r];
            ps += __shfl_xor(ps, 1, 32);
            ps += __shfl_xor(ps, 2, 32);
            ps += __shfl_xor(ps, 4, 32);
            ps += __shfl_xor(ps, 8, 32);
            rsum[r] = rsum[r] * alpha[r] + ps;
        }

#pragma unroll
        for (int r = 0; r < 8; ++r) {
            const int nr = r + (lo ? 0 : 8);
            Ps[wv][nr][mrow]      = (_Float16)p0[r];
            Ps[wv][nr][16 + mrow] = (_Float16)p1[r];
        }
        __syncthreads();

        const _Float16* pp = &Ps[wv][mrow][0];
        const v16h pf = lo ? make_frag(pp + 0, pp + 16) : make_frag(pp + 8, pp + 24);
#pragma unroll
        for (int hc = 0; hc < 8; ++hc) {
#pragma unroll
            for (int r = 0; r < 8; ++r) oacc[hc][r] *= alpha[r];
            const _Float16* vp = &Vs[cur][hc * 16 + mrow][lo ? 0 : 8];
            const v16h vf = make_frag(vp, vp + 16);
            oacc[hc] = __builtin_amdgcn_wmma_f32_16x16x32_f16(false, pf, false, vf,
                                                              (short)0, oacc[hc], false, false);
        }
        __syncthreads();    // all waves done reading buffer `cur`
        cur ^= 1;
    }

    float* Ob = Og + (size_t)b * HD * NPTS;
    float rinv[8];
#pragma unroll
    for (int r = 0; r < 8; ++r) rinv[r] = 1.f / rsum[r];
#pragma unroll
    for (int hc = 0; hc < 8; ++hc) {
        const int h = hc * 16 + mrow;
#pragma unroll
        for (int r = 0; r < 8; ++r) {
            const int n = n0 + wv * 16 + r + (lo ? 0 : 8);
            Ob[(size_t)h * NPTS + n] = oacc[hc][r] * rinv[r];
        }
    }
}

// ---------------------------------------------------------------------------
// 5) attention-pool score, softmax over N, weighted pool, final FC
// ---------------------------------------------------------------------------
__global__ void score_kernel(const float* __restrict__ A, const float* __restrict__ aw2,
                             const float* __restrict__ ab2, float* __restrict__ S) {
    const int b = blockIdx.y;
    const int n = blockIdx.x * 256 + threadIdx.x;
    const float* Ab = A + (size_t)b * 64 * NPTS;
    float s = ab2[0];
#pragma unroll
    for (int c = 0; c < 64; ++c) s += Ab[(size_t)c * NPTS + n] * aw2[c];
    S[(size_t)b * NPTS + n] = s;
}

__global__ void softmax_n_kernel(const float* __restrict__ S, float* __restrict__ Wt) {
    const int b = blockIdx.x;
    const float* Sb = S + (size_t)b * NPTS;
    __shared__ float red[256];
    float mx = -3.0e38f;
    for (int n = threadIdx.x; n < NPTS; n += 256) mx = fmaxf(mx, Sb[n]);
    red[threadIdx.x] = mx; __syncthreads();
    for (int st = 128; st > 0; st >>= 1) {
        if (threadIdx.x < st) red[threadIdx.x] = fmaxf(red[threadIdx.x], red[threadIdx.x + st]);
        __syncthreads();
    }
    mx = red[0]; __syncthreads();
    float sm = 0.f;
    for (int n = threadIdx.x; n < NPTS; n += 256) sm += __expf(Sb[n] - mx);
    red[threadIdx.x] = sm; __syncthreads();
    for (int st = 128; st > 0; st >>= 1) {
        if (threadIdx.x < st) red[threadIdx.x] += red[threadIdx.x + st];
        __syncthreads();
    }
    const float inv = 1.f / red[0];
    for (int n = threadIdx.x; n < NPTS; n += 256)
        Wt[(size_t)b * NPTS + n] = __expf(Sb[n] - mx) * inv;
}

__global__ void pool_kernel(const float* __restrict__ H, const float* __restrict__ Wt,
                            float* __restrict__ P) {
    const int c = blockIdx.x, b = blockIdx.y;
    const float* Hr = H + (size_t)b * 256 * NPTS + (size_t)c * NPTS;
    const float* Wb = Wt + (size_t)b * NPTS;
    __shared__ float red[256];
    float s = 0.f;
    for (int n = threadIdx.x; n < NPTS; n += 256) s += Hr[n] * Wb[n];
    red[threadIdx.x] = s; __syncthreads();
    for (int st = 128; st > 0; st >>= 1) {
        if (threadIdx.x < st) red[threadIdx.x] += red[threadIdx.x + st];
        __syncthreads();
    }
    if (threadIdx.x == 0) P[(size_t)b * 256 + c] = red[0];
}

__global__ void fc_kernel(const float* __restrict__ P, const float* __restrict__ fcw,
                          const float* __restrict__ fcb, float* __restrict__ out) {
    const int b = blockIdx.x, o = threadIdx.x;
    const float* Pb = P + (size_t)b * 256;
    float s = fcb[o];
    for (int c = 0; c < 256; ++c) s += Pb[c] * fcw[(size_t)o * 256 + c];
    out[(size_t)b * 256 + o] = fmaxf(s, 0.f);
}

// ---------------------------------------------------------------------------
extern "C" void kernel_launch(void* const* d_in, const int* in_sizes, int n_in,
                              void* d_out, int out_size, void* d_ws, size_t ws_size,
                              hipStream_t stream) {
    (void)in_sizes; (void)n_in; (void)out_size; (void)ws_size;
    const float* x   = (const float*)d_in[0];
    const float* w1  = (const float*)d_in[1];  const float* b1  = (const float*)d_in[2];
    const float* g1  = (const float*)d_in[3];  const float* be1 = (const float*)d_in[4];
    const float* w2  = (const float*)d_in[5];  const float* b2  = (const float*)d_in[6];
    const float* g2  = (const float*)d_in[7];  const float* be2 = (const float*)d_in[8];
    const float* w3  = (const float*)d_in[9];  const float* b3  = (const float*)d_in[10];
    const float* g3  = (const float*)d_in[11]; const float* be3 = (const float*)d_in[12];
    const float* qw  = (const float*)d_in[13]; const float* qb  = (const float*)d_in[14];
    const float* kw  = (const float*)d_in[15]; const float* kb  = (const float*)d_in[16];
    const float* vw  = (const float*)d_in[17]; const float* vb  = (const float*)d_in[18];
    const float* fw  = (const float*)d_in[19]; const float* fb  = (const float*)d_in[20];
    const float* aw1 = (const float*)d_in[21]; const float* ab1 = (const float*)d_in[22];
    const float* aw2 = (const float*)d_in[23]; const float* ab2 = (const float*)d_in[24];
    const float* fcw = (const float*)d_in[25]; const float* fcb = (const float*)d_in[26];

    float* ws = (float*)d_ws;
    const size_t BN = (size_t)BATCH * NPTS;
    size_t off = 0;
    float* feat = ws + off; off += 18 * BN;
    float* h1   = ws + off; off += 64 * BN;
    float* h2   = ws + off; off += 128 * BN;
    float* h3   = ws + off; off += 256 * BN;
    float* obuf = ws + off; off += 128 * BN;
    float* h4   = ws + off; off += 256 * BN;
    float* ab   = ws + off; off += 64 * BN;
    _Float16* qt = (_Float16*)(ws + off); off += 64 * BN;   // 128*BN halves
    _Float16* kt = (_Float16*)(ws + off); off += 64 * BN;
    _Float16* vh = (_Float16*)(ws + off); off += 64 * BN;
    float* sc   = ws + off; off += BN;
    float* wt   = ws + off; off += BN;
    float* pld  = ws + off; off += (size_t)BATCH * 256;
    float* scale = ws + off; off += 256;
    float* shift = ws + off; off += 256;

    knn_feat_kernel<<<dim3(NPTS / 256, BATCH), 256, 0, stream>>>(x, feat);

    // conv1: 18 -> 64, BN, ReLU
    gemm_kernel<18, 64><<<dim3(NPTS / 64, 2, BATCH), 128, 0, stream>>>(w1, feat, b1, nullptr, h1, nullptr, 0, 0);
    bn_stats_kernel<<<64, 256, 0, stream>>>(h1, g1, be1, 64, scale, shift);
    bn_apply_relu_kernel<<<(unsigned)((size_t)BATCH * 64 * NPTS / 256), 256, 0, stream>>>(h1, scale, shift, 64);

    // conv2: 64 -> 128, BN, ReLU
    gemm_kernel<64, 128><<<dim3(NPTS / 64, 4, BATCH), 128, 0, stream>>>(w2, h1, b2, nullptr, h2, nullptr, 0, 0);
    bn_stats_kernel<<<128, 256, 0, stream>>>(h2, g2, be2, 128, scale, shift);
    bn_apply_relu_kernel<<<(unsigned)((size_t)BATCH * 128 * NPTS / 256), 256, 0, stream>>>(h2, scale, shift, 128);

    // conv3: 128 -> 256, BN, ReLU
    gemm_kernel<128, 256><<<dim3(NPTS / 64, 8, BATCH), 128, 0, stream>>>(w3, h2, b3, nullptr, h3, nullptr, 0, 0);
    bn_stats_kernel<<<256, 256, 0, stream>>>(h3, g3, be3, 256, scale, shift);
    bn_apply_relu_kernel<<<(unsigned)((size_t)BATCH * 256 * NPTS / 256), 256, 0, stream>>>(h3, scale, shift, 256);

    // Q, K (transposed f16 [n][h]) and V (f16 [h][n]) projections: 256 -> 128
    gemm_kernel<256, 128><<<dim3(NPTS / 64, 4, BATCH), 128, 0, stream>>>(qw, h3, qb, nullptr, nullptr, qt, 1, 0);
    gemm_kernel<256, 128><<<dim3(NPTS / 64, 4, BATCH), 128, 0, stream>>>(kw, h3, kb, nullptr, nullptr, kt, 1, 0);
    gemm_kernel<256, 128><<<dim3(NPTS / 64, 4, BATCH), 128, 0, stream>>>(vw, h3, vb, nullptr, nullptr, vh, 2, 0);

    // global attention (flash style, double-buffered async LDS staging)
    flash_attn_kernel<<<dim3(NPTS / 64, BATCH), 128, 0, stream>>>(qt, kt, vh, obuf);

    // f-projection 128 -> 256 with residual add of h3
    gemm_kernel<128, 256><<<dim3(NPTS / 64, 8, BATCH), 128, 0, stream>>>(fw, obuf, fb, h3, h4, nullptr, 0, 0);

    // attention-pool MLP: 256 -> 64 (ReLU), then score 64 -> 1
    gemm_kernel<256, 64><<<dim3(NPTS / 64, 2, BATCH), 128, 0, stream>>>(aw1, h4, ab1, nullptr, ab, nullptr, 0, 1);
    score_kernel<<<dim3(NPTS / 256, BATCH), 256, 0, stream>>>(ab, aw2, ab2, sc);
    softmax_n_kernel<<<BATCH, 256, 0, stream>>>(sc, wt);
    pool_kernel<<<dim3(256, BATCH), 256, 0, stream>>>(h4, wt, pld);
    fc_kernel<<<BATCH, 256, 0, stream>>>(pld, fcw, fcb, (float*)d_out);
}
